// TextMaskedMultiheadSelfAttention_73830487818265
// MI455X (gfx1250) — matrix operand order repaired and checked by reading, not compiled, not run on hardware
//
#include <hip/hip_runtime.h>
#include <hip/hip_bf16.h>

typedef __attribute__((ext_vector_type(16))) _Float16 v16h;
typedef __attribute__((ext_vector_type(8)))  _Float16 v8h;
typedef __attribute__((ext_vector_type(4)))  _Float16 v4h;
typedef __attribute__((ext_vector_type(8)))  float    v8f;

#define WMMA_F16(a, b, c) \
  __builtin_amdgcn_wmma_f32_16x16x32_f16(false, (a), false, (b), (short)0, (c), false, false)

constexpr int Bsz = 2, T = 2048, D = 1024, H = 16, HD = 64;
constexpr int M = Bsz * T;                       // 4096 token rows
constexpr float QSCALE = 0.022097086912079608f;  // 1/sqrt(2048)

// LDS byte-offset of a __shared__ object (generic -> AS(3) -> 32-bit offset)
typedef __attribute__((address_space(3))) void lds_void_t;
__device__ __forceinline__ unsigned lds_off(void* p) {
  return (unsigned)(unsigned long long)(lds_void_t*)p;
}

// Async copy 16 bytes global -> LDS (VDST = LDS byte addr, VADDR = 64-bit global).
// Tracked by ASYNCcnt (CDNA5 ISA).
__device__ __forceinline__ void async_b128(unsigned lofs, const void* g) {
  asm volatile("global_load_async_to_lds_b128 %0, %1, off"
               :: "v"(lofs), "v"(g) : "memory");
}
__device__ __forceinline__ void wait_async0() {
  asm volatile("s_wait_asynccnt 0x0" ::: "memory");
}

// ---------------------------------------------------------------------------
// Kernel 1: x f32 -> f16 (row-major [M, D])
// ---------------------------------------------------------------------------
__global__ __launch_bounds__(256) void prep_x_kernel(const float* __restrict__ x,
                                                     _Float16* __restrict__ xh) {
  int idx = (blockIdx.x * 256 + threadIdx.x) * 4;
  float4 f = *(const float4*)(x + idx);
  v4h h;
  h[0] = (_Float16)f.x; h[1] = (_Float16)f.y;
  h[2] = (_Float16)f.z; h[3] = (_Float16)f.w;
  *(v4h*)(xh + idx) = h;
}

// ---------------------------------------------------------------------------
// Kernel 2: transpose + convert the 4 weight matrices: w[N,K] f32 -> wT[K,N] f16
// ---------------------------------------------------------------------------
__global__ __launch_bounds__(256) void prep_w_kernel(const float* __restrict__ wq,
                                                     const float* __restrict__ wk,
                                                     const float* __restrict__ wv,
                                                     const float* __restrict__ wo,
                                                     _Float16* __restrict__ wT) {
  __shared__ float tile[32][33];
  const float* w = (blockIdx.z == 0) ? wq : (blockIdx.z == 1) ? wk
                 : (blockIdx.z == 2) ? wv : wo;
  _Float16* out = wT + (size_t)blockIdx.z * D * D;
  const int tx = threadIdx.x, ty = threadIdx.y;
  const int kbase = blockIdx.x * 32, nbase = blockIdx.y * 32;
#pragma unroll
  for (int r = 0; r < 32; r += 8)
    tile[ty + r][tx] = w[(size_t)(nbase + ty + r) * D + kbase + tx];
  __syncthreads();
#pragma unroll
  for (int r = 0; r < 32; r += 8)
    out[(size_t)(kbase + ty + r) * D + nbase + tx] = (_Float16)tile[tx][ty + r];
}

// ---------------------------------------------------------------------------
// Kernel 3: fused QKV projection. grid=(M/128, D/64, 3), block=256 (8 waves).
// The 8 waves share one B tile (32K x 64N f16 = 4KB), double-buffered in LDS,
// filled by async global->LDS loads (one b128 chunk per thread per K-step).
// Wave w computes rows [blk*128 + 16w, +16); epilogue scatters Q/K/V layouts.
// ---------------------------------------------------------------------------
__global__ __launch_bounds__(256) void qkv_kernel(const _Float16* __restrict__ xh,
                                                  const _Float16* __restrict__ wT,
                                                  const float* __restrict__ bq,
                                                  const float* __restrict__ bk,
                                                  const float* __restrict__ bv,
                                                  _Float16* __restrict__ qh,
                                                  _Float16* __restrict__ kTh,
                                                  _Float16* __restrict__ vh) {
  __shared__ _Float16 bbuf[2][32 * 64];   // 2 x 4KB double buffer
  const int tid = threadIdx.x;
  const int lane = tid & 31;
  const int wave = tid >> 5;
  const int hi = lane >> 4, lo = lane & 15;
  const int m0 = blockIdx.x * 128 + wave * 16;
  const int n0 = blockIdx.y * 64;
  const int z = blockIdx.z;
  const _Float16* wm = wT + (size_t)z * D * D;
  const float* bias = (z == 0) ? bq : (z == 1) ? bk : bv;

  // async-fill assignment: 32 rows x 128B = 256 chunks of 16B, one per thread
  const int frow = tid >> 3;            // K row within tile (0..31)
  const int fc8 = tid & 7;              // 16B chunk within row
  const _Float16* gsrc = wm + (size_t)frow * D + n0 + fc8 * 8;
  const unsigned lbase = lds_off(&bbuf[0][0]);
  const unsigned lchunk = (unsigned)tid * 16u;

  v8f c[4];
#pragma unroll
  for (int j = 0; j < 4; ++j)
#pragma unroll
    for (int i = 0; i < 8; ++i) c[j][i] = 0.0f;

  // prologue: fill buffer 0 with K rows [0,32)
  async_b128(lbase + lchunk, gsrc);
  wait_async0();
  __syncthreads();

  const _Float16* arow = xh + (size_t)(m0 + lo) * D;
  int cur = 0;
  for (int k0 = 0; k0 < D; k0 += 32) {
    // prefetch next B tile into the other buffer
    if (k0 + 32 < D)
      async_b128(lbase + (unsigned)(1 - cur) * 4096u + lchunk,
                 gsrc + (size_t)(k0 + 32) * D);

    // A operand: lane = M row; K elements {ko..ko+7, 16+ko..16+ko+7}, ko=8*hi
    v8h p0 = *(const v8h*)(arow + k0 + hi * 8);
    v8h p1 = *(const v8h*)(arow + k0 + 16 + hi * 8);
    v16h a;
#pragma unroll
    for (int e = 0; e < 8; ++e) { a[e] = p0[e]; a[e + 8] = p1[e]; }

    // B operand from LDS: lane = N column, K elements {ko..ko+7, 16+ko..16+ko+7}
    const _Float16* brow = &bbuf[cur][lo];
#pragma unroll
    for (int j = 0; j < 4; ++j) {
      v16h b;
#pragma unroll
      for (int e = 0; e < 8; ++e) {
        b[e] = brow[(hi * 8 + e) * 64 + 16 * j];
        b[e + 8] = brow[(16 + hi * 8 + e) * 64 + 16 * j];
      }
      c[j] = WMMA_F16(a, b, c[j]);
    }

    wait_async0();
    __syncthreads();
    cur ^= 1;
  }

  const float scale = (z == 0) ? QSCALE : 1.0f;
#pragma unroll
  for (int j = 0; j < 4; ++j) {
    const int n = n0 + 16 * j + lo;
    const float bn = bias[n];
    const int hh = n >> 6, dd = n & 63;
#pragma unroll
    for (int i = 0; i < 8; ++i) {
      const int mm = m0 + i + hi * 8;
      const int bb = mm >> 11;          // / T
      const int tt = mm & (T - 1);
      const _Float16 val = (_Float16)((c[j][i] + bn) * scale);
      if (z == 0)      qh [(((size_t)(bb * H + hh)) * T + tt) * HD + dd] = val;
      else if (z == 1) kTh[(((size_t)(bb * H + hh)) * HD + dd) * T + tt] = val;
      else             vh [(((size_t)(bb * H + hh)) * T + tt) * HD + dd] = val;
    }
  }
}

// ---------------------------------------------------------------------------
// Kernel 4: flash-style attention. grid=(T/16, H, B), block=32 (one wave).
// 16-query tile streams 32-key blocks: scores via 4 WMMA, online softmax,
// P transposed D-layout -> A-layout through LDS, P*V via 4 WMMA.
// ---------------------------------------------------------------------------
__global__ __launch_bounds__(32) void attn_kernel(const _Float16* __restrict__ qh,
                                                  const _Float16* __restrict__ kTh,
                                                  const _Float16* __restrict__ vh,
                                                  _Float16* __restrict__ ctx) {
  __shared__ _Float16 pl[16][40];     // padded rows: 80B stride, 16B aligned
  const int lane = threadIdx.x;
  const int hi = lane >> 4, lo = lane & 15;
  const int q0 = blockIdx.x * 16;
  const int h = blockIdx.y;
  const int b = blockIdx.z;
  const size_t bh = (size_t)b * H + h;
  const _Float16* qp = qh + bh * (size_t)T * HD;
  const _Float16* kp = kTh + bh * (size_t)HD * T;
  const _Float16* vp = vh + bh * (size_t)T * HD;

  // Q tile A-operands (Q already scaled by 1/sqrt(T)): K-dim = hd = 64 -> 2 ops
  v16h aq0, aq1;
  {
    const _Float16* qrow = qp + (size_t)(q0 + lo) * HD;
    v8h x0 = *(const v8h*)(qrow + hi * 8);
    v8h x1 = *(const v8h*)(qrow + 16 + hi * 8);
    v8h x2 = *(const v8h*)(qrow + 32 + hi * 8);
    v8h x3 = *(const v8h*)(qrow + 48 + hi * 8);
#pragma unroll
    for (int e = 0; e < 8; ++e) {
      aq0[e] = x0[e]; aq0[e + 8] = x1[e];
      aq1[e] = x2[e]; aq1[e + 8] = x3[e];
    }
  }

  v8f acc[4];
  float mrow[8], lrow[8];
#pragma unroll
  for (int i = 0; i < 8; ++i) { mrow[i] = -1e30f; lrow[i] = 0.0f; }
#pragma unroll
  for (int j = 0; j < 4; ++j)
#pragma unroll
    for (int i = 0; i < 8; ++i) acc[j][i] = 0.0f;

  for (int kb = 0; kb < T; kb += 32) {
    // K^T B-operands: lane = key column, d elements in table order
    v16h bk0, bk1, bk2, bk3;
#pragma unroll
    for (int e = 0; e < 8; ++e) {
      const size_t d0 = (size_t)(hi * 8 + e), d1 = (size_t)(16 + hi * 8 + e);
      bk0[e] = kp[d0 * T + kb + lo];             bk0[e + 8] = kp[d1 * T + kb + lo];
      bk1[e] = kp[(32 + d0) * T + kb + lo];      bk1[e + 8] = kp[(32 + d1) * T + kb + lo];
      bk2[e] = kp[d0 * T + kb + 16 + lo];        bk2[e + 8] = kp[d1 * T + kb + 16 + lo];
      bk3[e] = kp[(32 + d0) * T + kb + 16 + lo]; bk3[e + 8] = kp[(32 + d1) * T + kb + 16 + lo];
    }

    v8f z8;
#pragma unroll
    for (int i = 0; i < 8; ++i) z8[i] = 0.0f;
    v8f s_lo = WMMA_F16(aq0, bk0, z8);
    s_lo = WMMA_F16(aq1, bk1, s_lo);
    v8f s_hi = WMMA_F16(aq0, bk2, z8);
    s_hi = WMMA_F16(aq1, bk3, s_hi);

    // online softmax; D-layout row = i + 8*hi, reductions inside 16-lane halves
#pragma unroll
    for (int i = 0; i < 8; ++i) {
      float v0 = s_lo[i], v1 = s_hi[i];
      float mx = fmaxf(v0, v1);
      mx = fmaxf(mx, __shfl_xor(mx, 1, 16));
      mx = fmaxf(mx, __shfl_xor(mx, 2, 16));
      mx = fmaxf(mx, __shfl_xor(mx, 4, 16));
      mx = fmaxf(mx, __shfl_xor(mx, 8, 16));
      float newm = fmaxf(mrow[i], mx);
      float alpha = __expf(mrow[i] - newm);
      float p0 = __expf(v0 - newm);
      float p1 = __expf(v1 - newm);
      float rs = p0 + p1;
      rs += __shfl_xor(rs, 1, 16);
      rs += __shfl_xor(rs, 2, 16);
      rs += __shfl_xor(rs, 4, 16);
      rs += __shfl_xor(rs, 8, 16);
      lrow[i] = lrow[i] * alpha + rs;
      mrow[i] = newm;
#pragma unroll
      for (int j = 0; j < 4; ++j) acc[j][i] *= alpha;
      pl[i + hi * 8][lo] = (_Float16)p0;
      pl[i + hi * 8][16 + lo] = (_Float16)p1;
    }
    asm volatile("s_wait_dscnt 0" ::: "memory");

    // reload P in A-layout: lane = query row, K = key column
    v16h ap;
    {
      v8h t0 = *(const v8h*)(&pl[lo][hi * 8]);
      v8h t1 = *(const v8h*)(&pl[lo][16 + hi * 8]);
#pragma unroll
      for (int e = 0; e < 8; ++e) { ap[e] = t0[e]; ap[e + 8] = t1[e]; }
    }

    // V B-operands: lane = d column, key elements in table order
#pragma unroll
    for (int j = 0; j < 4; ++j) {
      v16h bv16;
#pragma unroll
      for (int e = 0; e < 8; ++e) {
        bv16[e] = vp[(size_t)(kb + hi * 8 + e) * HD + 16 * j + lo];
        bv16[e + 8] = vp[(size_t)(kb + 16 + hi * 8 + e) * HD + 16 * j + lo];
      }
      acc[j] = WMMA_F16(ap, bv16, acc[j]);
    }
  }

  // normalize and store ctx as [B*T, D] f16 with column = h*64 + d
#pragma unroll
  for (int i = 0; i < 8; ++i) {
    const float inv = 1.0f / lrow[i];
    const int tt = q0 + i + hi * 8;
    const size_t row = ((size_t)b * T + tt) * (size_t)D;
#pragma unroll
    for (int j = 0; j < 4; ++j) {
      const int n = h * HD + 16 * j + lo;
      ctx[row + n] = (_Float16)(acc[j][i] * inv);
    }
  }
}

// ---------------------------------------------------------------------------
// Kernel 5: output projection ctx @ wo^T + bo -> f32 out. grid=(M/128, D/64),
// block=256; same async double-buffered shared-B scheme as qkv_kernel.
// ---------------------------------------------------------------------------
__global__ __launch_bounds__(256) void oproj_kernel(const _Float16* __restrict__ ctx,
                                                    const _Float16* __restrict__ wTo,
                                                    const float* __restrict__ bo,
                                                    float* __restrict__ out) {
  __shared__ _Float16 bbuf[2][32 * 64];
  const int tid = threadIdx.x;
  const int lane = tid & 31;
  const int wave = tid >> 5;
  const int hi = lane >> 4, lo = lane & 15;
  const int m0 = blockIdx.x * 128 + wave * 16;
  const int n0 = blockIdx.y * 64;

  const int frow = tid >> 3;
  const int fc8 = tid & 7;
  const _Float16* gsrc = wTo + (size_t)frow * D + n0 + fc8 * 8;
  const unsigned lbase = lds_off(&bbuf[0][0]);
  const unsigned lchunk = (unsigned)tid * 16u;

  v8f c[4];
#pragma unroll
  for (int j = 0; j < 4; ++j)
#pragma unroll
    for (int i = 0; i < 8; ++i) c[j][i] = 0.0f;

  async_b128(lbase + lchunk, gsrc);
  wait_async0();
  __syncthreads();

  const _Float16* arow = ctx + (size_t)(m0 + lo) * D;
  int cur = 0;
  for (int k0 = 0; k0 < D; k0 += 32) {
    if (k0 + 32 < D)
      async_b128(lbase + (unsigned)(1 - cur) * 4096u + lchunk,
                 gsrc + (size_t)(k0 + 32) * D);

    v8h p0 = *(const v8h*)(arow + k0 + hi * 8);
    v8h p1 = *(const v8h*)(arow + k0 + 16 + hi * 8);
    v16h a;
#pragma unroll
    for (int e = 0; e < 8; ++e) { a[e] = p0[e]; a[e + 8] = p1[e]; }

    const _Float16* brow = &bbuf[cur][lo];
#pragma unroll
    for (int j = 0; j < 4; ++j) {
      v16h b;
#pragma unroll
      for (int e = 0; e < 8; ++e) {
        b[e] = brow[(hi * 8 + e) * 64 + 16 * j];
        b[e + 8] = brow[(16 + hi * 8 + e) * 64 + 16 * j];
      }
      c[j] = WMMA_F16(a, b, c[j]);
    }

    wait_async0();
    __syncthreads();
    cur ^= 1;
  }

#pragma unroll
  for (int j = 0; j < 4; ++j) {
    const int n = n0 + 16 * j + lo;
    const float bn = bo[n];
#pragma unroll
    for (int i = 0; i < 8; ++i) {
      const int mm = m0 + i + hi * 8;
      out[(size_t)mm * D + n] = c[j][i] + bn;
    }
  }
}

// ---------------------------------------------------------------------------
extern "C" void kernel_launch(void* const* d_in, const int* in_sizes, int n_in,
                              void* d_out, int out_size, void* d_ws, size_t ws_size,
                              hipStream_t stream) {
  const float* x  = (const float*)d_in[0];
  const float* wq = (const float*)d_in[1];
  const float* bq = (const float*)d_in[2];
  const float* wk = (const float*)d_in[3];
  const float* bk = (const float*)d_in[4];
  const float* wv = (const float*)d_in[5];
  const float* bv = (const float*)d_in[6];
  const float* wo = (const float*)d_in[7];
  const float* bo = (const float*)d_in[8];
  float* out = (float*)d_out;

  char* ws = (char*)d_ws;
  const size_t MB = (size_t)1 << 20;
  _Float16* xh  = (_Float16*)(ws + 0 * MB);    // [M, D]          8 MiB
  _Float16* wT  = (_Float16*)(ws + 8 * MB);    // 4x [D, D]       8 MiB
  _Float16* qh  = (_Float16*)(ws + 16 * MB);   // [B,H,T,hd]      8 MiB
  _Float16* kTh = (_Float16*)(ws + 24 * MB);   // [B,H,hd,T]      8 MiB
  _Float16* vh  = (_Float16*)(ws + 32 * MB);   // [B,H,T,hd]      8 MiB
  _Float16* ch  = (_Float16*)(ws + 40 * MB);   // [M, D]          8 MiB

  prep_x_kernel<<<dim3(M * D / 4 / 256), dim3(256), 0, stream>>>(x, xh);
  prep_w_kernel<<<dim3(D / 32, D / 32, 4), dim3(32, 8), 0, stream>>>(wq, wk, wv, wo, wT);
  qkv_kernel<<<dim3(M / 128, D / 64, 3), dim3(256), 0, stream>>>(xh, wT, bq, bk, bv, qh, kTh, vh);
  attn_kernel<<<dim3(T / 16, H, Bsz), dim3(32), 0, stream>>>(qh, kTh, vh, ch);
  oproj_kernel<<<dim3(M / 128, D / 64), dim3(256), 0, stream>>>(ch, wT + (size_t)3 * D * D, bo, out);
}
